// Mamba2Block_38972533244319
// MI455X (gfx1250) — compile-verified, matrix operand order checked by reading, not run on hardware
//
#include <hip/hip_runtime.h>

// ---- problem dims (match reference) ----
#define DIMX   1024
#define STATE  64
#define KW     4
#define INNER  2048
#define BATCHN 2
#define SEQL   2048
#define NTOK   (BATCHN * SEQL)
#define EPSR   1e-6f

typedef __bf16 bf16_t;
typedef __attribute__((ext_vector_type(16))) __bf16 v16bf;
typedef __attribute__((ext_vector_type(8)))  __bf16 v8bf;
typedef __attribute__((ext_vector_type(8)))  float  v8f;
typedef __attribute__((ext_vector_type(4)))  int    v4i;

// gfx1250 async global->LDS copy (ASYNCcnt-tracked); fallback: load + ds_store.
#if defined(__gfx1250__) && __has_builtin(__builtin_amdgcn_global_load_async_to_lds_b128)
#define USE_ASYNC_LDS 1
#else
#define USE_ASYNC_LDS 0
#endif

__device__ __forceinline__ void copy16_to_lds(const bf16_t* gp, bf16_t* lp) {
#if USE_ASYNC_LDS
    __builtin_amdgcn_global_load_async_to_lds_b128(
        (__attribute__((address_space(1))) v4i*)gp,
        (__attribute__((address_space(3))) v4i*)lp, 0, 0);
#else
    *(v8bf*)lp = *(const v8bf*)gp;
#endif
}

__device__ __forceinline__ void wait_async_copies() {
#if USE_ASYNC_LDS
#if __has_builtin(__builtin_amdgcn_s_wait_asynccnt)
    __builtin_amdgcn_s_wait_asynccnt(0);
#else
    asm volatile("s_wait_asynccnt 0x0" ::: "memory");
#endif
#endif
}

// ----------------------------------------------------------------------------
// f32 -> bf16 elementwise convert (weights + activations)
// ----------------------------------------------------------------------------
__global__ void cvt_bf16_kernel(const float* __restrict__ src,
                                bf16_t* __restrict__ dst, int n) {
    int i = blockIdx.x * blockDim.x + threadIdx.x;
    if (i < n) dst[i] = (bf16_t)src[i];
}

// ----------------------------------------------------------------------------
// RMSNorm: one block per token, write bf16 normalized activations
// ----------------------------------------------------------------------------
__global__ void __launch_bounds__(256)
rmsnorm_kernel(const float* __restrict__ x, const float* __restrict__ w,
               bf16_t* __restrict__ out) {
    int tok = blockIdx.x;
    const float* xr = x + (size_t)tok * DIMX;
    float ss = 0.f;
    for (int i = threadIdx.x; i < DIMX; i += 256) { float v = xr[i]; ss += v * v; }
    for (int o = 16; o; o >>= 1) ss += __shfl_xor(ss, o, 32);
    __shared__ float sred[8];
    int lane = threadIdx.x & 31, wv = threadIdx.x >> 5;
    if (lane == 0) sred[wv] = ss;
    __syncthreads();
    if (wv == 0) {
        float t = (lane < 8) ? sred[lane] : 0.f;
        for (int o = 4; o; o >>= 1) t += __shfl_xor(t, o, 32);
        if (lane == 0) sred[0] = t;
    }
    __syncthreads();
    float rs = rsqrtf(sred[0] * (1.f / DIMX) + EPSR);
    bf16_t* orow = out + (size_t)tok * DIMX;
    for (int i = threadIdx.x; i < DIMX; i += 256)
        orow[i] = (bf16_t)(xr[i] * rs * w[i]);
}

// ----------------------------------------------------------------------------
// Blocked WMMA bf16 GEMM:  C[M,N] = A[M,K] @ B[N,K]^T  (+ epilogue)
//   Block: 256 thr (8 waves), block tile 128x128.
//   Wave grid 2x4; each wave computes 64x32 = 4x2 WMMA accumulators.
//   K-slices of A/B (128x32 bf16, row stride padded to 40 elems) are staged
//   in double-buffered LDS via async global->LDS copies; compute on slice k
//   overlaps the copy of slice k+1 (ASYNCcnt + barrier pipeline).
//   mode 0: plain | mode 1: softplus(val+bias[col])+1e-4 | mode 2: +resid
// Fragment layout per CDNA5 ISA 7.12.2 (16-bit A/B striping):
//   lanes 0-15:  row = lane,     K elems {k..k+7,  k+16..k+23}
//   lanes 16-31: row = lane-16,  K elems {k+8..k+15, k+24..k+31}
// C layout: VGPR v, lanes 0-15 -> (M=v, N=lane); lanes 16-31 -> (M=v+8, N=lane-16)
// ----------------------------------------------------------------------------
#define LDS_STRIDE 40   // 32 K-elems + 8 pad (keeps 16B alignment, spreads banks)

__global__ void __launch_bounds__(256)
wmma_gemm_kernel(const bf16_t* __restrict__ A, const bf16_t* __restrict__ B,
                 float* __restrict__ C, int M, int N, int K,
                 const float* __restrict__ bias,
                 const float* __restrict__ resid, int mode) {
    __shared__ bf16_t s_a[2][128 * LDS_STRIDE];
    __shared__ bf16_t s_b[2][128 * LDS_STRIDE];

    const int lane = threadIdx.x & 31;
    const int wavi = threadIdx.x >> 5;       // 0..7
    const int wm   = wavi >> 2;              // 0..1 -> 64-row slab
    const int wn   = wavi & 3;               // 0..3 -> 32-col slab
    const int half = lane >> 4;              // 0/1
    const int r    = lane & 15;
    const int off  = half * 8;

    const int blockM = blockIdx.y * 128;
    const int blockN = blockIdx.x * 128;

    union Frag { v16bf v; v8bf h[2]; };
    v8f acc[4][2] = {};

    const int nK = K >> 5;

    auto stage = [&](int ks) {
        const int kbuf = ks & 1;
        const size_t k0 = (size_t)ks * 32;
        // A slice: 128 rows x 32 K = 512 16-byte chunks
        for (int c = threadIdx.x; c < 512; c += 256) {
            int row = c >> 2, part = (c & 3) * 8;
            copy16_to_lds(A + (size_t)(blockM + row) * K + k0 + part,
                          &s_a[kbuf][row * LDS_STRIDE + part]);
        }
        // B slice
        for (int c = threadIdx.x; c < 512; c += 256) {
            int row = c >> 2, part = (c & 3) * 8;
            copy16_to_lds(B + (size_t)(blockN + row) * K + k0 + part,
                          &s_b[kbuf][row * LDS_STRIDE + part]);
        }
    };

    stage(0);
    for (int ks = 0; ks < nK; ++ks) {
        wait_async_copies();
        __syncthreads();                       // slice ks resident; prior reads done
        if (ks + 1 < nK) stage(ks + 1);        // overlap copy of next slice

        const int kbuf = ks & 1;
        const bf16_t* sa = s_a[kbuf];
        const bf16_t* sb = s_b[kbuf];

        Frag af[4], bfr[2];
#pragma unroll
        for (int mi = 0; mi < 4; ++mi) {
            int row = (wm * 64 + mi * 16 + r) * LDS_STRIDE;
            af[mi].h[0] = *(const v8bf*)&sa[row + off];
            af[mi].h[1] = *(const v8bf*)&sa[row + off + 16];
        }
#pragma unroll
        for (int ni = 0; ni < 2; ++ni) {
            int col = (wn * 32 + ni * 16 + r) * LDS_STRIDE;
            bfr[ni].h[0] = *(const v8bf*)&sb[col + off];
            bfr[ni].h[1] = *(const v8bf*)&sb[col + off + 16];
        }
#pragma unroll
        for (int mi = 0; mi < 4; ++mi)
#pragma unroll
            for (int ni = 0; ni < 2; ++ni)
                acc[mi][ni] = __builtin_amdgcn_wmma_f32_16x16x32_bf16(
                    false, af[mi].v, false, bfr[ni].v, (short)0,
                    acc[mi][ni], false, false);
    }

    // epilogue
#pragma unroll
    for (int ni = 0; ni < 2; ++ni) {
        const int colg = blockN + wn * 32 + ni * 16 + r;
        const float bv = (mode == 1) ? bias[colg] : 0.f;
#pragma unroll
        for (int mi = 0; mi < 4; ++mi) {
#pragma unroll
            for (int v = 0; v < 8; ++v) {
                int   rowg = blockM + wm * 64 + mi * 16 + v + half * 8;
                float val  = acc[mi][ni][v];
                if (mode == 1) {
                    val += bv;
                    val = (val > 20.f) ? val : log1pf(__expf(val));  // softplus
                    val += 1e-4f;
                } else if (mode == 2) {
                    val += resid[(size_t)rowg * N + colg];
                }
                C[(size_t)rowg * N + colg] = val;
            }
        }
    }
}

// ----------------------------------------------------------------------------
// Causal depthwise conv (KW=4) + bias + SiLU over xc half of xz
// ----------------------------------------------------------------------------
__global__ void conv_silu_kernel(const float* __restrict__ xz,
                                 const float* __restrict__ w,
                                 const float* __restrict__ b,
                                 float* __restrict__ xcf,
                                 bf16_t* __restrict__ xcb) {
    int idx = blockIdx.x * blockDim.x + threadIdx.x;
    if (idx >= NTOK * INNER) return;
    int d   = idx % INNER;
    int tok = idx / INNER;
    int l   = tok % SEQL;
    float acc = b[d];
#pragma unroll
    for (int j = 0; j < KW; ++j) {
        int li = l + j - (KW - 1);
        if (li >= 0)
            acc += w[d * KW + j] * xz[(size_t)(tok + j - (KW - 1)) * (2 * INNER) + d];
    }
    float s = acc / (1.f + __expf(-acc));   // SiLU
    xcf[idx] = s;
    xcb[idx] = (bf16_t)s;
}

// ----------------------------------------------------------------------------
// SSM scan: 1 wave per (b, d) channel; 2 states/lane (STATE=64, wave32).
// Bx/Cx come interleaved from one [NTOK,128] buffer (cols 0..63 = Bx, 64..127 = Cx).
// ----------------------------------------------------------------------------
__global__ void __launch_bounds__(256)
ssm_scan_kernel(const float* __restrict__ dt, const float* __restrict__ bxcx,
                const float* __restrict__ xc,
                const float* __restrict__ A_log, const float* __restrict__ Dp,
                float* __restrict__ y) {
    __shared__ float s_bx[32][STATE];
    __shared__ float s_cx[32][STATE];
    __shared__ float s_dt[32][8];
    __shared__ float s_x [32][8];

    int lane = threadIdx.x & 31;
    int wave = threadIdx.x >> 5;
    int b    = blockIdx.y;
    int d    = blockIdx.x * 8 + wave;
    int n0   = lane, n1 = lane + 32;

    float A0 = -__expf(A_log[(size_t)d * STATE + n0]);
    float A1 = -__expf(A_log[(size_t)d * STATE + n1]);
    float Dv = Dp[d];
    float h0 = 0.f, h1 = 0.f;
    size_t tokBase = (size_t)b * SEQL;

    for (int t0 = 0; t0 < SEQL; t0 += 32) {
        for (int i = threadIdx.x; i < 32 * STATE; i += 256) {
            int tt = i >> 6, n = i & 63;
            size_t g = (tokBase + t0 + tt) * 128 + n;
            s_bx[tt][n] = bxcx[g];
            s_cx[tt][n] = bxcx[g + 64];
        }
        for (int i = threadIdx.x; i < 32 * 8; i += 256) {
            int tt = i >> 3, w = i & 7;
            size_t g = (tokBase + t0 + tt) * INNER + (size_t)blockIdx.x * 8 + w;
            s_dt[tt][w] = dt[g];
            s_x [tt][w] = xc[g];
        }
        __syncthreads();
#pragma unroll 4
        for (int tt = 0; tt < 32; ++tt) {
            float dtv = s_dt[tt][wave];
            float xv  = s_x [tt][wave];
            float dx  = dtv * xv;
            h0 = __expf(dtv * A0) * h0 + dx * s_bx[tt][n0];
            h1 = __expf(dtv * A1) * h1 + dx * s_bx[tt][n1];
            float c = h0 * s_cx[tt][n0] + h1 * s_cx[tt][n1];
            for (int o = 16; o; o >>= 1) c += __shfl_xor(c, o, 32);
            if (lane == 0)
                y[(tokBase + t0 + tt) * INNER + d] = c + Dv * xv;
        }
        __syncthreads();
    }
}

// ----------------------------------------------------------------------------
// y *= SiLU(z); emit bf16 for out_proj
// ----------------------------------------------------------------------------
__global__ void gate_silu_kernel(const float* __restrict__ y,
                                 const float* __restrict__ xz,
                                 bf16_t* __restrict__ yb) {
    int idx = blockIdx.x * blockDim.x + threadIdx.x;
    if (idx >= NTOK * INNER) return;
    int d   = idx % INNER;
    int tok = idx / INNER;
    float z = xz[(size_t)tok * (2 * INNER) + INNER + d];
    float g = z / (1.f + __expf(-z));
    yb[idx] = (bf16_t)(y[idx] * g);
}

// ----------------------------------------------------------------------------
extern "C" void kernel_launch(void* const* d_in, const int* in_sizes, int n_in,
                              void* d_out, int out_size, void* d_ws, size_t ws_size,
                              hipStream_t stream) {
    const float* x          = (const float*)d_in[0];
    const float* in_proj_w  = (const float*)d_in[1];
    const float* conv_w     = (const float*)d_in[2];
    const float* conv_b     = (const float*)d_in[3];
    const float* A_log      = (const float*)d_in[4];
    const float* B_w        = (const float*)d_in[5];
    const float* C_w        = (const float*)d_in[6];
    const float* D_param    = (const float*)d_in[7];
    const float* dt_w       = (const float*)d_in[8];
    const float* dt_b       = (const float*)d_in[9];
    const float* out_proj_w = (const float*)d_in[10];
    const float* norm_w     = (const float*)d_in[11];

    char* ws = (char*)d_ws;
    size_t off = 0;
    auto carve = [&](size_t bytes) {
        char* p = ws + off;
        off = (off + bytes + 255) & ~(size_t)255;
        return p;
    };
    bf16_t* wInB  = (bf16_t*)carve((size_t)2 * INNER * DIMX * 2);
    bf16_t* wDtB  = (bf16_t*)carve((size_t)INNER * INNER * 2);
    bf16_t* wOutB = (bf16_t*)carve((size_t)DIMX * INNER * 2);
    bf16_t* wBCB  = (bf16_t*)carve((size_t)2 * STATE * INNER * 2);  // [Bw;Cw] = [128, INNER]
    bf16_t* xnB   = (bf16_t*)carve((size_t)NTOK * DIMX * 2);
    float*  xzF   = (float*) carve((size_t)NTOK * 2 * INNER * 4);
    float*  xcF   = (float*) carve((size_t)NTOK * INNER * 4);
    bf16_t* xcB   = (bf16_t*)carve((size_t)NTOK * INNER * 2);
    float*  dtF   = (float*) carve((size_t)NTOK * INNER * 4);
    float*  bcF   = (float*) carve((size_t)NTOK * 128 * 4);         // [Bx|Cx] interleaved
    float*  yF    = (float*) carve((size_t)NTOK * INNER * 4);
    bf16_t* yB    = (bf16_t*)carve((size_t)NTOK * INNER * 2);
    (void)ws_size; (void)in_sizes; (void)n_in; (void)out_size;

    auto cvt = [&](const float* s, bf16_t* dst, int n) {
        cvt_bf16_kernel<<<(n + 255) / 256, 256, 0, stream>>>(s, dst, n);
    };
    cvt(in_proj_w,  wInB,  2 * INNER * DIMX);
    cvt(dt_w,       wDtB,  INNER * INNER);
    cvt(out_proj_w, wOutB, DIMX * INNER);
    cvt(B_w,        wBCB,                 STATE * INNER);
    cvt(C_w,        wBCB + STATE * INNER, STATE * INNER);

    // 1) RMSNorm -> bf16 activations
    rmsnorm_kernel<<<NTOK, 256, 0, stream>>>(x, norm_w, xnB);

    // 2) in_proj: [NTOK, 2*INNER] = xn @ W_in^T
    wmma_gemm_kernel<<<dim3((2 * INNER) / 128, NTOK / 128), 256, 0, stream>>>(
        xnB, wInB, xzF, NTOK, 2 * INNER, DIMX, nullptr, nullptr, 0);

    // 3) causal depthwise conv + SiLU on xc half
    conv_silu_kernel<<<(NTOK * INNER + 255) / 256, 256, 0, stream>>>(
        xzF, conv_w, conv_b, xcF, xcB);

    // 4) dt = softplus(xc @ dt_w^T + dt_b) + 1e-4
    wmma_gemm_kernel<<<dim3(INNER / 128, NTOK / 128), 256, 0, stream>>>(
        xcB, wDtB, dtF, NTOK, INNER, INNER, dt_b, nullptr, 1);

    // 5) fused Bx|Cx projection (N = 128)
    wmma_gemm_kernel<<<dim3(1, NTOK / 128), 256, 0, stream>>>(
        xcB, wBCB, bcF, NTOK, 128, INNER, nullptr, nullptr, 0);

    // 6) sequential SSM scan (1 wave per channel, LDS-tiled)
    ssm_scan_kernel<<<dim3(INNER / 8, BATCHN), 256, 0, stream>>>(
        dtF, bcF, xcF, A_log, D_param, yF);

    // 7) gate with SiLU(z)
    gate_silu_kernel<<<(NTOK * INNER + 255) / 256, 256, 0, stream>>>(yF, xzF, yB);

    // 8) out_proj + residual -> d_out (fp32)
    wmma_gemm_kernel<<<dim3(DIMX / 128, NTOK / 128), 256, 0, stream>>>(
        yB, wOutB, (float*)d_out, NTOK, DIMX, INNER, nullptr, x, 2);
}
